// FocalToVoxelNeXtBridge_45561013076206
// MI455X (gfx1250) — matrix operand order repaired and checked by reading, not compiled, not run on hardware
//
#include <hip/hip_runtime.h>
#include <hip/hip_bf16.h>

typedef __attribute__((ext_vector_type(16))) _Float16 v16h;
typedef __attribute__((ext_vector_type(8)))  float    v8f;
typedef __attribute__((ext_vector_type(4)))  unsigned int v4u;

union Frag { v16h h; v4u u[2]; };

// Problem constants (from reference)
constexpr int BV  = 2;
constexpr int NV  = 400000;
constexpr int CIN = 192;
constexpr int CC  = 128;
constexpr int YV  = 512;
constexpr int XV  = 512;
constexpr float EPS1F = 1e-5f;
constexpr float EPS2F = 1e-3f;

constexpr size_t DENSE_ELEMS = (size_t)BV * YV * XV * CC;   // 67,108,864
constexpr size_t DENSE_BYTES = DENSE_ELEMS * sizeof(float); // 268,435,456
constexpr size_t ACTIVE_BYTES = (size_t)BV * YV * XV;       // 524,288

// ---------------------------------------------------------------------------
// Kernel 1: f = relu(bn1(vf @ W + b)) scatter-added into dense grid.
// 256 threads = 8 waves; wave w computes rows [blk*128 + w*16, +16) x 128 cols.
// ---------------------------------------------------------------------------
__global__ __launch_bounds__(256) void proj_scatter_kernel(
    const float* __restrict__ vf, const int* __restrict__ coords,
    const float* __restrict__ Wp, const float* __restrict__ bp,
    const float* __restrict__ g1, const float* __restrict__ be1,
    const float* __restrict__ m1, const float* __restrict__ v1,
    float* __restrict__ dense, unsigned char* __restrict__ act)
{
    extern __shared__ char smem[];
    _Float16* Wl   = (_Float16*)smem;                // [192][128] f16 = 49152 B
    int*      cell = (int*)(smem + CIN * CC * 2);    // [128]

    const int tid  = threadIdx.x;
    const int lane = tid & 31;
    const int w    = tid >> 5;
    const int rowbase = blockIdx.x * 128;

    // Stage projection weights fp32 -> f16 in LDS (shared by all 8 waves).
    for (int e = tid; e < CIN * CC; e += 256)
        Wl[e] = (_Float16)Wp[e];

    // Stage scatter targets; mark active cells (z coord intentionally dropped).
    if (tid < 128) {
        const int gr = rowbase + tid;
        const int bi = coords[gr * 4 + 0];
        const int yi = coords[gr * 4 + 2];
        const int xi = coords[gr * 4 + 3];
        const int cl = (bi * YV + yi) * XV + xi;
        cell[tid] = cl;
        act[cl] = (unsigned char)1;
    }
    __syncthreads();

    v8f acc[8] = {};

    // A fragment addressing (ISA 7.12.2, 16-bit A 16x32):
    // lanes 0-15: row=lane,  K = {0..7, 16..23}
    // lanes16-31: row=lane&15, K = {8..15, 24..31}
    const int mrow = rowbase + (w << 4) + (lane & 15);
    const int koff = (lane >= 16) ? 8 : 0;

    for (int kc = 0; kc < CIN / 32; ++kc) {
        const float* ap = vf + (size_t)mrow * CIN + kc * 32 + koff;
        v16h av;
        #pragma unroll
        for (int j = 0; j < 8; ++j) {
            av[j]     = (_Float16)ap[j];        // K chunk 0..7 (+koff)
            av[8 + j] = (_Float16)ap[16 + j];   // K chunk 16..23 (+koff)
        }
        const int krow = kc * 32 + lane;        // B: lane&15 = N col, halves = K (as A)

        // Load phase: issue all 128 16-bit LDS loads back-to-back so DS returns
        // overlap with the WMMA chain (partial dscnt waits instead of 0).
        Frag b[8];
        #pragma unroll
        for (int nt = 0; nt < 8; ++nt) {
            #pragma unroll
            for (int j = 0; j < 8; ++j) {
                b[nt].h[j]     = Wl[(kc * 32 + koff + j) * CC + nt * 16 + (lane & 15)];
                b[nt].h[8 + j] = Wl[(kc * 32 + koff + 16 + j) * CC + nt * 16 + (lane & 15)];
            }
        }
        // Compute phase: 8 independent WMMAs.
        #pragma unroll
        for (int nt = 0; nt < 8; ++nt) {
            acc[nt] = __builtin_amdgcn_wmma_f32_16x16x32_f16(
                false, av, false, b[nt].h, (short)0, acc[nt], false, false);
        }
    }

    // Epilogue: BN1 + ReLU + scatter-add (f32 atomics into dense NHWC grid).
    // D layout: VGPR r, lanes 0-15 -> M=r, lanes 16-31 -> M=r+8; N = lane&15.
    const int mhi = (lane >> 4) * 8;
    #pragma unroll
    for (int nt = 0; nt < 8; ++nt) {
        const int n = nt * 16 + (lane & 15);
        const float sc = g1[n] * rsqrtf(v1[n] + EPS1F);
        const float sh = (bp[n] - m1[n]) * sc + be1[n];
        #pragma unroll
        for (int r = 0; r < 8; ++r) {
            const int lm = (w << 4) + mhi + r;  // local row 0..127
            float val = fmaxf(acc[nt][r] * sc + sh, 0.f);
            atomicAdd(dense + (size_t)cell[lm] * CC + n, val);
        }
    }
}

// ---------------------------------------------------------------------------
// Kernel 2: 3x3 SAME conv (implicit GEMM) + bias + BN2 + ReLU + active mask.
// One workgroup = 128 consecutive output pixels in one row x all 128 channels.
// Wave w handles pixels [w*16, w*16+16).
// LDS: input patch [3][130][128] f16 (99840 B) + per-tap weights [128][128]
// f16 (32768 B) restaged each tap -> 132608 B (fits 320 KB WGP LDS).
// ---------------------------------------------------------------------------
__global__ __launch_bounds__(256) void conv_bn_relu_kernel(
    const float* __restrict__ dense, const unsigned char* __restrict__ act,
    const float* __restrict__ cw, const float* __restrict__ cb,
    const float* __restrict__ g2, const float* __restrict__ be2,
    const float* __restrict__ m2, const float* __restrict__ v2,
    float* __restrict__ out)
{
    extern __shared__ char smem[];
    _Float16* in_t = (_Float16*)smem;                          // [3][130][128]
    _Float16* wt   = (_Float16*)(smem + 3 * 130 * CC * 2);     // [128][128]

    const int tid  = threadIdx.x;
    const int lane = tid & 31;
    const int w    = tid >> 5;

    const int bid = blockIdx.x;
    const int x0 = (bid & 3) * 128;      // 4 segments per row
    const int y  = (bid >> 2) & (YV - 1);
    const int b  = bid >> 11;

    // Stage 3x130x128 input patch (zero-padded halo), fp32 -> f16.
    for (int e = tid; e < 3 * 130 * CC; e += 256) {
        const int iy  = e / (130 * CC);
        const int rem = e - iy * 130 * CC;
        const int ix  = rem >> 7;
        const int c   = rem & (CC - 1);
        const int gy  = y - 1 + iy;
        const int gx  = x0 - 1 + ix;
        float fv = 0.f;
        if (gy >= 0 && gy < YV && gx >= 0 && gx < XV)
            fv = dense[((size_t)(b * YV + gy) * XV + gx) * CC + c];
        in_t[e] = (_Float16)fv;
    }

    v8f acc[8] = {};
    const int px   = (w << 4) + (lane & 15);   // local output pixel for A row
    const int koff = (lane >= 16) ? 8 : 0;

    for (int t = 0; t < 9; ++t) {
        __syncthreads();  // also guards initial in_t staging at t==0
        const float* wsrc = cw + (size_t)t * CC * CC;   // conv_w[ky][kx][ci][co]
        for (int e = tid; e < CC * CC; e += 256)
            wt[e] = (_Float16)wsrc[e];

        // Prefetch next tap's weight block into cache while this tap computes
        // (global_prefetch_b8; 256 thr x 2 lines covers the 64 KB region).
        if (t < 8) {
            const float* wnext = cw + (size_t)(t + 1) * CC * CC;
            __builtin_prefetch(wnext + (size_t)tid * 32, 0, 1);
            __builtin_prefetch(wnext + (size_t)(tid + 256) * 32, 0, 1);
        }
        __syncthreads();

        const int ky = t / 3;
        const int kx = t - ky * 3;
        const _Float16* arow = in_t + ((size_t)(ky * 130 + (px + kx))) * CC;

        #pragma unroll
        for (int kc = 0; kc < 4; ++kc) {          // K = 128 in 4 chunks of 32
            Frag a;
            a.u[0] = *(const v4u*)(arow + kc * 32 + koff);
            a.u[1] = *(const v4u*)(arow + kc * 32 + koff + 16);
            const int krow = kc * 32 + lane;

            // Load phase: all B fragments for this k-chunk.
            Frag bb[8];
            #pragma unroll
            for (int nt = 0; nt < 8; ++nt) {
                #pragma unroll
                for (int j = 0; j < 8; ++j) {
                    bb[nt].h[j]     = wt[(kc * 32 + koff + j) * CC + nt * 16 + (lane & 15)];
                    bb[nt].h[8 + j] = wt[(kc * 32 + koff + 16 + j) * CC + nt * 16 + (lane & 15)];
                }
            }
            // Compute phase: 8 independent WMMAs.
            #pragma unroll
            for (int nt = 0; nt < 8; ++nt) {
                acc[nt] = __builtin_amdgcn_wmma_f32_16x16x32_f16(
                    false, a.h, false, bb[nt].h, (short)0, acc[nt], false, false);
            }
        }
    }

    // Epilogue: + conv bias, BN2, ReLU, zero inactive cells, store fp32.
    const int mhi = (lane >> 4) * 8;
    #pragma unroll
    for (int nt = 0; nt < 8; ++nt) {
        const int n = nt * 16 + (lane & 15);
        const float sc = g2[n] * rsqrtf(v2[n] + EPS2F);
        const float sh = (cb[n] - m2[n]) * sc + be2[n];
        #pragma unroll
        for (int r = 0; r < 8; ++r) {
            const int gx = x0 + (w << 4) + mhi + r;
            const size_t pix = (size_t)(b * YV + y) * XV + gx;
            float val = fmaxf(acc[nt][r] * sc + sh, 0.f);
            if (!act[pix]) val = 0.f;
            out[pix * CC + n] = val;
        }
    }
}

// ---------------------------------------------------------------------------
extern "C" void kernel_launch(void* const* d_in, const int* in_sizes, int n_in,
                              void* d_out, int out_size, void* d_ws, size_t ws_size,
                              hipStream_t stream) {
    (void)in_sizes; (void)n_in; (void)out_size; (void)ws_size;

    const float* vf     = (const float*)d_in[0];
    const int*   coords = (const int*)  d_in[1];
    const float* Wp     = (const float*)d_in[2];
    const float* bp     = (const float*)d_in[3];
    const float* g1     = (const float*)d_in[4];
    const float* be1    = (const float*)d_in[5];
    const float* m1     = (const float*)d_in[6];
    const float* v1     = (const float*)d_in[7];
    const float* cw     = (const float*)d_in[8];
    const float* cb     = (const float*)d_in[9];
    const float* g2     = (const float*)d_in[10];
    const float* be2    = (const float*)d_in[11];
    const float* m2     = (const float*)d_in[12];
    const float* v2     = (const float*)d_in[13];

    float* dense       = (float*)d_ws;
    unsigned char* act = (unsigned char*)d_ws + DENSE_BYTES;
    float* out         = (float*)d_out;

    // Zero dense accumulator + active mask (graph-capturable memset node).
    hipMemsetAsync(d_ws, 0, DENSE_BYTES + ACTIVE_BYTES, stream);

    // Projection GEMM + BN1 + ReLU + scatter. 400000 / 128 = 3125 exactly.
    const size_t smem1 = (size_t)CIN * CC * 2 + 128 * sizeof(int); // 49664 B
    proj_scatter_kernel<<<dim3(NV / 128), dim3(256), smem1, stream>>>(
        vf, coords, Wp, bp, g1, be1, m1, v1, dense, act);

    // Conv 3x3 + BN2 + ReLU + mask. (512/128) * 512 * 2 = 4096 workgroups.
    const size_t smem2 = (size_t)3 * 130 * CC * 2 + (size_t)CC * CC * 2; // 132608 B
    conv_bn_relu_kernel<<<dim3((XV / 128) * YV * BV), dim3(256), smem2, stream>>>(
        dense, act, cw, cb, g2, be2, m2, v2, out);
}
